// PointAttentionBlock_70909910057218
// MI455X (gfx1250) — matrix operand order repaired and checked by reading, not compiled, not run on hardware
//
#include <hip/hip_runtime.h>

// Problem constants (match reference)
#define B_    8
#define C_    256
#define N_    4096      // 64*64
#define CQK_  32        // C/8
#define OTOT  320       // 32 (q) + 32 (k) + 256 (v)
#define WPB   4         // waves per block in flash kernel

typedef __attribute__((ext_vector_type(16))) _Float16 v16h;
typedef __attribute__((ext_vector_type(8)))  float    v8f;

union AFrag { v16h v; uint4 u[2]; };

// ---------------------------------------------------------------------------
// Kernel 0a: x [B,C,N] f32  ->  xT [B,N,C] f16  (K-dim contiguous for WMMA B)
// ---------------------------------------------------------------------------
__global__ void cvt_x(const float* __restrict__ x, _Float16* __restrict__ xT) {
  int idx = blockIdx.x * 256 + threadIdx.x;      // exactly B*N*C threads
  int c  = idx & (C_ - 1);
  int bn = idx >> 8;                             // b*N + n
  int n  = bn & (N_ - 1);
  int b  = bn >> 12;
  xT[idx] = (_Float16)x[(b * C_ + c) * N_ + n];
}

// ---------------------------------------------------------------------------
// Kernel 0b: concat weights -> Wcat [320,256] f16, bcat [320] f32
// ---------------------------------------------------------------------------
__global__ void cvt_w(const float* __restrict__ wq, const float* __restrict__ bq,
                      const float* __restrict__ wk, const float* __restrict__ bk,
                      const float* __restrict__ wv, const float* __restrict__ bv,
                      _Float16* __restrict__ Wcat, float* __restrict__ bcat) {
  int idx = blockIdx.x * 256 + threadIdx.x;
  if (idx < OTOT * C_) {
    int o = idx / C_, c = idx % C_;
    float v;
    if (o < 32)       v = wq[o * C_ + c];
    else if (o < 64)  v = wk[(o - 32) * C_ + c];
    else              v = wv[(o - 64) * C_ + c];
    Wcat[idx] = (_Float16)v;
  } else if (idx < OTOT * C_ + OTOT) {
    int o = idx - OTOT * C_;
    bcat[o] = (o < 32) ? bq[o] : (o < 64) ? bk[o - 32] : bv[o - 64];
  }
}

// ---------------------------------------------------------------------------
// Kernel 1: fused q/k/v projection.  One wave per 16x16 tile of [320 x 4096].
// D(M=o, N=p) = sum_c Wcat[o,c] * x[c,p]  via 8x v_wmma_f32_16x16x32_f16.
// Outputs: qT,kT [B,N,32] f16 (d contiguous), vh [B,C,N] f16 (j contiguous).
// ---------------------------------------------------------------------------
__global__ void __launch_bounds__(32)
qkv_gemm(const _Float16* __restrict__ xT, const _Float16* __restrict__ W,
         const float* __restrict__ bias,
         _Float16* __restrict__ qT, _Float16* __restrict__ kT,
         _Float16* __restrict__ vh) {
  const int lane = threadIdx.x & 31;
  const int half = lane >> 4, l16 = lane & 15;
  const int p0 = blockIdx.x * 16;
  const int o0 = blockIdx.y * 16;
  const int b  = blockIdx.z;

  v8f acc = {};
  // A layout: lanes0-15 hold K = 0..7 & 16..23; lanes16-31 hold K = 8..15 & 24..31
  const _Float16* wrow = W + (o0 + l16) * C_ + half * 8;
  // B layout: lanes0-15 hold K = 0..7 & 16..23; lanes16-31 hold K = 8..15 & 24..31
  const _Float16* xrow = xT + (size_t)(b * N_ + p0 + l16) * C_ + half * 8;
#pragma unroll
  for (int c0 = 0; c0 < C_; c0 += 32) {
    AFrag a, bm;
    a.u[0]  = *(const uint4*)(wrow + c0);
    a.u[1]  = *(const uint4*)(wrow + c0 + 16);
    bm.u[0] = *(const uint4*)(xrow + c0);
    bm.u[1] = *(const uint4*)(xrow + c0 + 16);
    acc = __builtin_amdgcn_wmma_f32_16x16x32_f16(false, a.v, false, bm.v,
                                                 (short)0, acc, false, false);
  }
  // D layout: reg r -> M = r + half*8 ; lane -> N = l16
  const int p = p0 + l16;
#pragma unroll
  for (int r = 0; r < 8; ++r) {
    int o = o0 + half * 8 + r;
    float val = acc[r] + bias[o];
    if (o < 32)
      qT[(size_t)(b * N_ + p) * CQK_ + o] = (_Float16)val;
    else if (o < 64)
      kT[(size_t)(b * N_ + p) * CQK_ + (o - 32)] = (_Float16)val;
    else
      vh[(size_t)(b * C_ + (o - 64)) * N_ + p] = (_Float16)val;
  }
}

// ---------------------------------------------------------------------------
// Kernel 2: two-pass flash attention, 32 query rows per wave.
// Two 16-row tiles share every K and V fragment (2x arithmetic intensity:
// each V B-fragment feeds two PV WMMAs), halving WGP$/L2 traffic per FLOP.
// Accumulators: 2 x 16 x v8f = 256 VGPRs -- fine under wave32's 1024 budget;
// 32 independent WMMA chains provide the ILP so 1-2 waves/SIMD suffices.
//
// Pass 1 (max):  S = Q*K^T, lane-local elementwise running max only.
// Pass 2 (sum+PV): recompute S, p = exp(s - m) unnormalized, elementwise
//   row-sum, P -> LDS D->A transpose, 32 PV WMMAs per 32 columns.
// Epilogue: one shuffle-reduce of row sums; out = alpha*acc/l + x.
// ---------------------------------------------------------------------------
__global__ void __launch_bounds__(WPB * 32)
flash(const _Float16* __restrict__ qT, const _Float16* __restrict__ kT,
      const _Float16* __restrict__ vh, const float* __restrict__ x,
      const float* __restrict__ alpha, float* __restrict__ out) {
  // padded stride 40 f16: rows stay 16B aligned, bank-spread on 64 banks
  __shared__ __align__(16) _Float16 plds[WPB][32 * 40];
  const int tid  = threadIdx.x;
  const int w    = tid >> 5, lane = tid & 31;
  const int half = lane >> 4, l16 = lane & 15;
  const int b    = blockIdx.y;
  const int i0   = (blockIdx.x * WPB + w) * 32;    // 32 query rows per wave

  // Q A-fragments for both 16-row tiles (K = d = 0..31 is one WMMA K step)
  AFrag qa0, qa1;
  {
    const _Float16* qrow = qT + (size_t)(b * N_ + i0 + l16) * CQK_ + half * 8;
    qa0.u[0] = *(const uint4*)qrow;
    qa0.u[1] = *(const uint4*)(qrow + 16);
    const _Float16* qrow1 = qrow + 16 * CQK_;
    qa1.u[0] = *(const uint4*)qrow1;
    qa1.u[1] = *(const uint4*)(qrow1 + 16);
  }

  const _Float16* kbase = kT + (size_t)b * N_ * CQK_;
  const _Float16* vbase = vh + (size_t)b * C_ * N_;
  _Float16* myp = &plds[w][0];
  const v8f z = {};

  // ---------------- Pass 1: row max (both tiles share K fragments) --------
  float mx0[8], mx1[8];
#pragma unroll
  for (int r = 0; r < 8; ++r) { mx0[r] = -1e30f; mx1[r] = -1e30f; }

  for (int j0 = 0; j0 < N_; j0 += 32) {
    AFrag ka, kb;
    const _Float16* kr0 = kbase + (size_t)(j0 + l16) * CQK_ + half * 8;
    const _Float16* kr1 = kr0 + 16 * CQK_;
    ka.u[0] = *(const uint4*)kr0;       ka.u[1] = *(const uint4*)(kr0 + 16);
    kb.u[0] = *(const uint4*)kr1;       kb.u[1] = *(const uint4*)(kr1 + 16);
    v8f s00 = __builtin_amdgcn_wmma_f32_16x16x32_f16(false, qa0.v, false, ka.v,
                                                     (short)0, z, false, false);
    v8f s01 = __builtin_amdgcn_wmma_f32_16x16x32_f16(false, qa0.v, false, kb.v,
                                                     (short)0, z, false, false);
    v8f s10 = __builtin_amdgcn_wmma_f32_16x16x32_f16(false, qa1.v, false, ka.v,
                                                     (short)0, z, false, false);
    v8f s11 = __builtin_amdgcn_wmma_f32_16x16x32_f16(false, qa1.v, false, kb.v,
                                                     (short)0, z, false, false);
#pragma unroll
    for (int r = 0; r < 8; ++r) {
      mx0[r] = fmaxf(mx0[r], fmaxf(s00[r], s01[r]));
      mx1[r] = fmaxf(mx1[r], fmaxf(s10[r], s11[r]));
    }
  }
  // One shuffle-reduce: D layout keeps row (r + half*8) inside a 16-lane
  // group, so xor masks 1,2,4,8 complete the row reduction.
  float m0[8], m1[8];
#pragma unroll
  for (int r = 0; r < 8; ++r) {
    float t = mx0[r];
    t = fmaxf(t, __shfl_xor(t, 1, 32));
    t = fmaxf(t, __shfl_xor(t, 2, 32));
    t = fmaxf(t, __shfl_xor(t, 4, 32));
    t = fmaxf(t, __shfl_xor(t, 8, 32));
    m0[r] = t;
    float u = mx1[r];
    u = fmaxf(u, __shfl_xor(u, 1, 32));
    u = fmaxf(u, __shfl_xor(u, 2, 32));
    u = fmaxf(u, __shfl_xor(u, 4, 32));
    u = fmaxf(u, __shfl_xor(u, 8, 32));
    m1[r] = u;
  }

  // ---------------- Pass 2: exp, row sum, P*V ----------------
  v8f acc0[16], acc1[16];
#pragma unroll
  for (int c = 0; c < 16; ++c) { acc0[c] = (v8f){}; acc1[c] = (v8f){}; }
  float ls0[8], ls1[8];
#pragma unroll
  for (int r = 0; r < 8; ++r) { ls0[r] = 0.f; ls1[r] = 0.f; }

  for (int j0 = 0; j0 < N_; j0 += 32) {
    AFrag ka, kb;
    const _Float16* kr0 = kbase + (size_t)(j0 + l16) * CQK_ + half * 8;
    const _Float16* kr1 = kr0 + 16 * CQK_;
    ka.u[0] = *(const uint4*)kr0;       ka.u[1] = *(const uint4*)(kr0 + 16);
    kb.u[0] = *(const uint4*)kr1;       kb.u[1] = *(const uint4*)(kr1 + 16);
    v8f s00 = __builtin_amdgcn_wmma_f32_16x16x32_f16(false, qa0.v, false, ka.v,
                                                     (short)0, z, false, false);
    v8f s01 = __builtin_amdgcn_wmma_f32_16x16x32_f16(false, qa0.v, false, kb.v,
                                                     (short)0, z, false, false);
    v8f s10 = __builtin_amdgcn_wmma_f32_16x16x32_f16(false, qa1.v, false, ka.v,
                                                     (short)0, z, false, false);
    v8f s11 = __builtin_amdgcn_wmma_f32_16x16x32_f16(false, qa1.v, false, kb.v,
                                                     (short)0, z, false, false);
    // p = exp(s - m), elementwise row-sum (normalization deferred; PV linear)
    float p00[8], p01[8], p10[8], p11[8];
#pragma unroll
    for (int r = 0; r < 8; ++r) {
      p00[r] = __expf(s00[r] - m0[r]);
      p01[r] = __expf(s01[r] - m0[r]);
      ls0[r] += p00[r] + p01[r];
      p10[r] = __expf(s10[r] - m1[r]);
      p11[r] = __expf(s11[r] - m1[r]);
      ls1[r] += p10[r] + p11[r];
    }
    // P: D layout -> LDS [row][col]; tile1 rows offset by 16
#pragma unroll
    for (int r = 0; r < 8; ++r) {
      int row = r + half * 8;
      myp[row * 40 + l16]             = (_Float16)p00[r];
      myp[row * 40 + 16 + l16]        = (_Float16)p01[r];
      myp[(16 + row) * 40 + l16]      = (_Float16)p10[r];
      myp[(16 + row) * 40 + 16 + l16] = (_Float16)p11[r];
    }
    asm volatile("s_wait_dscnt 0" ::: "memory");   // same-wave LDS RAW
    // Re-load P in A layout: M = l16; K halves 0..7/16..23 vs 8..15/24..31
    AFrag pa0, pa1;
    pa0.u[0] = *(const uint4*)&myp[l16 * 40 + half * 8];
    pa0.u[1] = *(const uint4*)&myp[l16 * 40 + 16 + half * 8];
    pa1.u[0] = *(const uint4*)&myp[(16 + l16) * 40 + half * 8];
    pa1.u[1] = *(const uint4*)&myp[(16 + l16) * 40 + 16 + half * 8];

    // Each V B-fragment (N = channel, K = j) feeds BOTH row tiles' WMMAs.
#pragma unroll
    for (int c = 0; c < 16; ++c) {
      AFrag vb;
      const _Float16* vr = vbase + (size_t)(c * 16 + l16) * N_ + j0 + half * 8;
      vb.u[0] = *(const uint4*)vr;
      vb.u[1] = *(const uint4*)(vr + 16);
      acc0[c] = __builtin_amdgcn_wmma_f32_16x16x32_f16(false, pa0.v, false, vb.v,
                                                       (short)0, acc0[c], false, false);
      acc1[c] = __builtin_amdgcn_wmma_f32_16x16x32_f16(false, pa1.v, false, vb.v,
                                                       (short)0, acc1[c], false, false);
    }
  }

  // ---------------- Epilogue: normalize + residual ----------------
  const float a0 = alpha[0];
  float inv0[8], inv1[8];
#pragma unroll
  for (int r = 0; r < 8; ++r) {
    float rs = ls0[r];
    rs += __shfl_xor(rs, 1, 32);
    rs += __shfl_xor(rs, 2, 32);
    rs += __shfl_xor(rs, 4, 32);
    rs += __shfl_xor(rs, 8, 32);
    inv0[r] = a0 / rs;
    float rt = ls1[r];
    rt += __shfl_xor(rt, 1, 32);
    rt += __shfl_xor(rt, 2, 32);
    rt += __shfl_xor(rt, 4, 32);
    rt += __shfl_xor(rt, 8, 32);
    inv1[r] = a0 / rt;
  }
#pragma unroll
  for (int c = 0; c < 16; ++c) {
    int ch = c * 16 + l16;
#pragma unroll
    for (int r = 0; r < 8; ++r) {
      int i = i0 + r + half * 8;
      size_t idx = (size_t)(b * C_ + ch) * N_ + i;
      out[idx]      = acc0[c][r] * inv0[r] + x[idx];
      out[idx + 16] = acc1[c][r] * inv1[r] + x[idx + 16];
    }
  }
}

// ---------------------------------------------------------------------------
// Workspace layout (bytes), all 16B aligned:
//   xT   f16 [B,N,C]   @ 0          (16,777,216)
//   Wcat f16 [320,256] @ 16777216   (163,840)
//   bcat f32 [320]     @ 16941056   (1,280)
//   qT   f16 [B,N,32]  @ 16942336   (2,097,152)
//   kT   f16 [B,N,32]  @ 19039488   (2,097,152)
//   vh   f16 [B,C,N]   @ 21136640   (16,777,216)   total ~37.9 MB
// ---------------------------------------------------------------------------
extern "C" void kernel_launch(void* const* d_in, const int* in_sizes, int n_in,
                              void* d_out, int out_size, void* d_ws, size_t ws_size,
                              hipStream_t stream) {
  const float* x     = (const float*)d_in[0];
  const float* wq    = (const float*)d_in[1];
  const float* bq    = (const float*)d_in[2];
  const float* wk    = (const float*)d_in[3];
  const float* bk    = (const float*)d_in[4];
  const float* wv    = (const float*)d_in[5];
  const float* bv    = (const float*)d_in[6];
  const float* alpha = (const float*)d_in[7];
  float* out = (float*)d_out;

  char* ws = (char*)d_ws;
  _Float16* xT = (_Float16*)(ws);
  _Float16* Wc = (_Float16*)(ws + 16777216);
  float*    bc = (float*)   (ws + 16941056);
  _Float16* qT = (_Float16*)(ws + 16942336);
  _Float16* kT = (_Float16*)(ws + 19039488);
  _Float16* vh = (_Float16*)(ws + 21136640);

  cvt_x<<<(B_ * N_ * C_) / 256, 256, 0, stream>>>(x, xT);
  cvt_w<<<(OTOT * C_ + OTOT + 255) / 256, 256, 0, stream>>>(wq, bq, wk, bk, wv, bv, Wc, bc);
  qkv_gemm<<<dim3(N_ / 16, OTOT / 16, B_), 32, 0, stream>>>(xT, Wc, bc, qT, kT, vh);
  flash<<<dim3(N_ / 32 / WPB, B_), WPB * 32, 0, stream>>>(qT, kT, vh, x, alpha, out);
}
